// SwinBlock_83640193122719
// MI455X (gfx1250) — compile-verified, matrix operand order checked by reading, not run on hardware
//
#include <hip/hip_runtime.h>

typedef __bf16 bf16_t;
typedef __bf16 v16bf __attribute__((ext_vector_type(16)));
typedef float  v8f   __attribute__((ext_vector_type(8)));

#define DIM     512
#define HEADS   16
#define HD      32
#define NWINTOK 64        // tokens per 8x8 window
#define NTOK    100352    // B * H * W = 32*56*56
#define NWIMG   49        // 7*7 windows per image
#define MLP_HID 2048
#define QKV_N   1536
#define BSTRIDE 40        // padded LDS row stride (elems) for B tiles
#define PSTRIDE 72        // padded LDS row stride (elems) for attention P tiles

// ---------------- wave reductions (wave32) ----------------
__device__ inline float wave_sum32(float v){
  #pragma unroll
  for (int m=16;m>=1;m>>=1) v += __shfl_xor(v,m,32);
  return v;
}
__device__ inline float half_max16(float v){
  #pragma unroll
  for (int m=8;m>=1;m>>=1) v = fmaxf(v, __shfl_xor(v,m,32));
  return v;
}
__device__ inline float half_sum16(float v){
  #pragma unroll
  for (int m=8;m>=1;m>>=1) v += __shfl_xor(v,m,32);
  return v;
}

// ---------------- WMMA fragment loader ----------------
// Loads a 16x32 bf16 A-operand fragment. Element (r,k) is read from
// base[r*rowStride + k*colStride].  Per CDNA5 ISA: lane(0..15)=row M,
// lanes 0-15 hold K {0..7,16..23}, lanes 16-31 hold K {8..15,24..31},
// VGPR d holds K pair {koff+2d, koff+2d+1} (d<4) / {16+koff+...} (d>=4).
// The B operand (32x16) uses the transposed view with the same loader.
__device__ inline v16bf load_frag(const bf16_t* __restrict__ base,
                                  int rowStride, int colStride){
  int lane = threadIdx.x & 31;
  int r    = lane & 15;
  int koff = (lane >> 4) << 3;   // 0 or 8
  const bf16_t* p = base + (size_t)r * rowStride;
  v16bf f;
  #pragma unroll
  for (int d=0; d<4; ++d){
    int k0 = koff + 2*d;
    int k1 = 16 + koff + 2*d;
    f[2*d]     = p[(size_t)(k0  )*colStride];
    f[2*d+1]   = p[(size_t)(k0+1)*colStride];
    f[8+2*d]   = p[(size_t)(k1  )*colStride];
    f[8+2*d+1] = p[(size_t)(k1+1)*colStride];
  }
  return f;
}

// ---------------- weight f32 -> bf16 ----------------
__global__ __launch_bounds__(256) void cvt_weights(
    const float* __restrict__ a, const float* __restrict__ b,
    const float* __restrict__ c, const float* __restrict__ d,
    bf16_t* __restrict__ o)
{
  size_t i = (size_t)blockIdx.x*256 + threadIdx.x;
  const size_t n0 = 786432;            // w_qkv  1536*512
  const size_t n1 = n0 + 262144;       // w_proj 512*512
  const size_t n2 = n1 + 1048576;      // w_fc1  2048*512
  const size_t n3 = n2 + 1048576;      // w_fc2  512*2048
  if      (i < n0) o[i] = (bf16_t)a[i];
  else if (i < n1) o[i] = (bf16_t)b[i-n0];
  else if (i < n2) o[i] = (bf16_t)c[i-n1];
  else if (i < n3) o[i] = (bf16_t)d[i-n2];
}

// ---------------- log-CPB positional bias table (16,64,64) ----------------
__global__ __launch_bounds__(256) void posbias_kernel(
    const float* __restrict__ w1, const float* __restrict__ b1,
    const float* __restrict__ w2, const float* __restrict__ b2,
    float* __restrict__ pb)
{
  int idx = blockIdx.x*256 + threadIdx.x;    // (n,m) pair
  if (idx >= 64*64) return;
  int n = idx >> 6, m = idx & 63;
  float dy = (float)((n>>3) - (m>>3));
  float dx = (float)((n&7)  - (m&7));
  float d0 = copysignf(log1pf(fabsf(dy)), dy);
  float d1 = copysignf(log1pf(fabsf(dx)), dx);
  float hbuf[32];
  #pragma unroll
  for (int j=0;j<32;++j)
    hbuf[j] = fmaxf(0.f, w1[2*j]*d0 + w1[2*j+1]*d1 + b1[j]);
  #pragma unroll
  for (int h=0;h<HEADS;++h){
    float s = b2[h];
    #pragma unroll
    for (int j=0;j<32;++j) s += w2[h*32+j]*hbuf[j];
    pb[(size_t)h*4096 + idx] = s;
  }
}

// ---------------- LayerNorm (one wave per token, 16 ch/lane) ----------------
// WINDOW=true: output token order is (window, n) with cyclic-shift gather.
template<bool WINDOW>
__global__ __launch_bounds__(256) void ln_kernel(
    const float* __restrict__ x, const float* __restrict__ g,
    const float* __restrict__ be, bf16_t* __restrict__ out)
{
  int t    = blockIdx.x*8 + (threadIdx.x >> 5);
  int lane = threadIdx.x & 31;
  size_t src;
  if (WINDOW){
    int win = t >> 6, n = t & 63;
    int b_ = win / NWIMG, wrem = win % NWIMG;
    int wh = wrem/7, ww = wrem%7, r = n>>3, c = n&7;
    int sh_ = (wh*8 + r + 4) % 56;               // roll(-4) gather
    int sw_ = (ww*8 + c + 4) % 56;
    src = (size_t)b_*3136 + sh_*56 + sw_;
  } else {
    src = (size_t)t;
  }
  const float* px = x + src*DIM;
  float v[16]; float s = 0.f;
  #pragma unroll
  for (int i=0;i<16;++i){ v[i] = px[lane + 32*i]; s += v[i]; }
  s = wave_sum32(s);
  float mean = s * (1.f/512.f);
  float q = 0.f;
  #pragma unroll
  for (int i=0;i<16;++i){ float d = v[i]-mean; q += d*d; }
  q = wave_sum32(q);
  float rstd = rsqrtf(q*(1.f/512.f) + 1e-5f);
  bf16_t* po = out + (size_t)t*DIM;
  #pragma unroll
  for (int i=0;i<16;++i){
    int ch = lane + 32*i;
    po[ch] = (bf16_t)((v[i]-mean)*rstd*g[ch] + be[ch]);
  }
}

// ---------------- WMMA GEMM: C[M,N] = A[M,K] @ Wt[N,K]^T + bias ----------------
// block 256 = 8 waves; wave -> 32x64 tile; block -> 256x64 tile.
// The 64x32 weight tile per K-step is shared by all 8 waves: it is staged
// cooperatively into LDS (one b128 per thread), double-buffered.
enum { EPI_BF16=0, EPI_BF16_GELU=1, EPI_PROJ_SCATTER=2, EPI_F32_RES=3 };

template<int EPI>
__global__ __launch_bounds__(256) void gemm_wmma(
    const bf16_t* __restrict__ A, const bf16_t* __restrict__ Wt,
    const float* __restrict__ bias, int K, int N,
    bf16_t* __restrict__ outb, float* __restrict__ outf,
    const float* __restrict__ res)
{
  __shared__ bf16_t Bs[2][64*BSTRIDE];
  int tid   = threadIdx.x;
  int wave  = tid >> 5;
  int lane  = tid & 31;
  int mbase = blockIdx.x*256 + wave*32;
  int nbase = blockIdx.y*64;

  // cooperative B-tile stage: tid -> (row r = tid/4, 16B chunk p = tid%4)
  int br = tid >> 2, bp = tid & 3;
  const bf16_t* bsrc_row = Wt + (size_t)(nbase + br)*K + bp*8;
  bf16_t*       bdst     = &Bs[0][0] + br*BSTRIDE + bp*8;

  *reinterpret_cast<int4*>(bdst) =
      *reinterpret_cast<const int4*>(bsrc_row);          // k0 = 0 into buf 0

  v8f acc[2][4] = {};
  for (int k0=0; k0<K; k0+=32){
    int buf = (k0 >> 5) & 1;
    __syncthreads();
    if (k0 + 32 < K)
      *reinterpret_cast<int4*>(bdst + (buf^1)*64*BSTRIDE) =
          *reinterpret_cast<const int4*>(bsrc_row + k0 + 32);
    v16bf a0 = load_frag(A + (size_t)(mbase     )*K + k0, K, 1);
    v16bf a1 = load_frag(A + (size_t)(mbase + 16)*K + k0, K, 1);
    #pragma unroll
    for (int j=0;j<4;++j){
      v16bf b = load_frag(&Bs[buf][0] + 16*j*BSTRIDE, BSTRIDE, 1);
      acc[0][j] = __builtin_amdgcn_wmma_f32_16x16x32_bf16(
                      false, a0, false, b, (short)0, acc[0][j], false, false);
      acc[1][j] = __builtin_amdgcn_wmma_f32_16x16x32_bf16(
                      false, a1, false, b, (short)0, acc[1][j], false, false);
    }
  }

  int rbase = (lane < 16) ? 0 : 8;
  #pragma unroll
  for (int mi=0; mi<2; ++mi){
    #pragma unroll
    for (int j=0;j<4;++j){
      #pragma unroll
      for (int i=0;i<8;++i){
        int gr = mbase + 16*mi + rbase + i;
        int gc = nbase + 16*j + (lane & 15);
        float v = acc[mi][j][i] + bias[gc];
        if (EPI == EPI_BF16){
          outb[(size_t)gr*N + gc] = (bf16_t)v;
        } else if (EPI == EPI_BF16_GELU){
          float gel = 0.5f*v*(1.f + erff(v*0.70710678118654752f));
          outb[(size_t)gr*N + gc] = (bf16_t)gel;
        } else if (EPI == EPI_PROJ_SCATTER){
          // window reverse + roll(+4) scatter, add shortcut x
          int win = gr >> 6, n = gr & 63;
          int b_ = win / NWIMG, wrem = win % NWIMG;
          int wh = wrem/7, ww = wrem%7, r = n>>3, c = n&7;
          int dh = (wh*8 + r + 4) % 56;
          int dw = (ww*8 + c + 4) % 56;
          size_t dst = ((size_t)b_*3136 + dh*56 + dw)*DIM + gc;
          outf[dst] = res[dst] + v;
        } else { // EPI_F32_RES
          size_t dst = (size_t)gr*N + gc;
          outf[dst] = res[dst] + v;
        }
      }
    }
  }
}

// ---------------- windowed attention: 1 block/window, 1 wave/head ----------------
__global__ __launch_bounds__(512) void attn_wmma(
    const bf16_t* __restrict__ qkv, const float* __restrict__ posb,
    bf16_t* __restrict__ outb)
{
  __shared__ bf16_t lds[16*16*PSTRIDE];  // per-wave 16x64 P tile, padded stride
  int wave = threadIdx.x >> 5;           // head
  int lane = threadIdx.x & 31;
  int h    = wave;
  int win  = blockIdx.x;
  size_t wbase = (size_t)win * NWINTOK;
  int wrem = win % NWIMG;
  int wh = wrem/7, ww = wrem%7;
  const float scale = 0.17677669529663687f;   // 1/sqrt(32)

  v16bf kf[4];
  #pragma unroll
  for (int jt=0;jt<4;++jt)
    kf[jt] = load_frag(qkv + (wbase + 16*jt)*QKV_N + DIM + h*HD, QKV_N, 1);
  v16bf vf[2][2];                         // V^T fragments: row=d, col=m
  #pragma unroll
  for (int kt=0;kt<2;++kt)
    #pragma unroll
    for (int jd=0;jd<2;++jd)
      vf[kt][jd] = load_frag(qkv + (wbase + 32*kt)*QKV_N + 2*DIM + h*HD + 16*jd,
                             1, QKV_N);

  bf16_t* myL = lds + wave*16*PSTRIDE;
  int rbase = (lane<16) ? 0 : 8;
  int mcol  = lane & 15;

  for (int it=0; it<4; ++it){
    v16bf qf = load_frag(qkv + (wbase + 16*it)*QKV_N + h*HD, QKV_N, 1);
    v8f a[4] = {};
    #pragma unroll
    for (int jt=0;jt<4;++jt)
      a[jt] = __builtin_amdgcn_wmma_f32_16x16x32_bf16(
                  false, qf, false, kf[jt], (short)0, a[jt], false, false);

    // logits = scale*QK^T + pos_bias + analytic shift mask
    #pragma unroll
    for (int jt=0;jt<4;++jt){
      #pragma unroll
      for (int i=0;i<8;++i){
        int n = 16*it + rbase + i;
        int m = 16*jt + mcol;
        float l = a[jt][i]*scale + posb[(size_t)h*4096 + n*64 + m];
        int grn = wh*8 + (n>>3), gcn = ww*8 + (n&7);
        int grm = wh*8 + (m>>3), gcm = ww*8 + (m&7);
        int bn = (grn<48?0:(grn<52?1:2))*3 + (gcn<48?0:(gcn<52?1:2));
        int bm = (grm<48?0:(grm<52?1:2))*3 + (gcm<48?0:(gcm<52?1:2));
        if (bn != bm) l -= 100.f;
        a[jt][i] = l;
      }
    }
    // row softmax over all 64 keys (registers + 16-lane half-wave shuffles)
    #pragma unroll
    for (int i=0;i<8;++i){
      float mx = -3.0e38f;
      #pragma unroll
      for (int jt=0;jt<4;++jt) mx = fmaxf(mx, a[jt][i]);
      mx = half_max16(mx);
      float s = 0.f;
      #pragma unroll
      for (int jt=0;jt<4;++jt){ float e = __expf(a[jt][i]-mx); a[jt][i]=e; s+=e; }
      s = half_sum16(s);
      float inv = 1.f/s;
      #pragma unroll
      for (int jt=0;jt<4;++jt)
        myL[(rbase+i)*PSTRIDE + 16*jt + mcol] = (bf16_t)(a[jt][i]*inv);
    }
    // out(16xHD) = P(16x64) @ V(64xHD); relayout P via LDS (same-wave DS is in-order)
    v16bf p0 = load_frag(myL,      PSTRIDE, 1);
    v16bf p1 = load_frag(myL + 32, PSTRIDE, 1);
    v8f o[2] = {};
    #pragma unroll
    for (int jd=0;jd<2;++jd){
      o[jd] = __builtin_amdgcn_wmma_f32_16x16x32_bf16(
                  false, p0, false, vf[0][jd], (short)0, o[jd], false, false);
      o[jd] = __builtin_amdgcn_wmma_f32_16x16x32_bf16(
                  false, p1, false, vf[1][jd], (short)0, o[jd], false, false);
    }
    #pragma unroll
    for (int jd=0;jd<2;++jd)
      #pragma unroll
      for (int i=0;i<8;++i){
        int n = 16*it + rbase + i;
        int d = 16*jd + mcol;
        outb[(wbase + n)*DIM + h*HD + d] = (bf16_t)o[jd][i];
      }
  }
}

// ---------------- launch ----------------
extern "C" void kernel_launch(void* const* d_in, const int* in_sizes, int n_in,
                              void* d_out, int out_size, void* d_ws, size_t ws_size,
                              hipStream_t stream)
{
  (void)in_sizes; (void)n_in; (void)out_size; (void)ws_size;
  const float* x      = (const float*)d_in[0];
  const float* g1     = (const float*)d_in[1];
  const float* be1    = (const float*)d_in[2];
  const float* w_qkv  = (const float*)d_in[3];
  const float* b_qkv  = (const float*)d_in[4];
  const float* pm_w1  = (const float*)d_in[5];
  const float* pm_b1  = (const float*)d_in[6];
  const float* pm_w2  = (const float*)d_in[7];
  const float* pm_b2  = (const float*)d_in[8];
  const float* w_proj = (const float*)d_in[9];
  const float* b_proj = (const float*)d_in[10];
  const float* g2     = (const float*)d_in[11];
  const float* be2    = (const float*)d_in[12];
  const float* w_fc1  = (const float*)d_in[13];
  const float* b_fc1  = (const float*)d_in[14];
  const float* w_fc2  = (const float*)d_in[15];
  const float* b_fc2  = (const float*)d_in[16];
  float* out = (float*)d_out;

  char* ws = (char*)d_ws;
  const size_t SXW  = (size_t)NTOK*DIM*2;       // windowed LN1 out (bf16)
  const size_t SBIG = (size_t)NTOK*MLP_HID*2;   // qkv (subset) / fc1 out (bf16)
  const size_t SATT = (size_t)NTOK*DIM*2;       // attn out / LN2 out (bf16)
  const size_t SX1  = (size_t)NTOK*DIM*4;       // x1 (f32)
  bf16_t* xw  = (bf16_t*)(ws);
  bf16_t* big = (bf16_t*)(ws + SXW);
  bf16_t* att = (bf16_t*)(ws + SXW + SBIG);
  float*  x1  = (float* )(ws + SXW + SBIG + SATT);
  bf16_t* wb  = (bf16_t*)(ws + SXW + SBIG + SATT + SX1);
  float*  pb  = (float* )(ws + SXW + SBIG + SATT + SX1
                              + (size_t)(786432+262144+1048576+1048576)*2);
  bf16_t* wb_qkv  = wb;
  bf16_t* wb_proj = wb + 786432;
  bf16_t* wb_fc1  = wb_proj + 262144;
  bf16_t* wb_fc2  = wb_fc1 + 1048576;

  cvt_weights<<<12288, 256, 0, stream>>>(w_qkv, w_proj, w_fc1, w_fc2, wb);
  posbias_kernel<<<16, 256, 0, stream>>>(pm_w1, pm_b1, pm_w2, pm_b2, pb);
  ln_kernel<true><<<NTOK/8, 256, 0, stream>>>(x, g1, be1, xw);
  gemm_wmma<EPI_BF16><<<dim3(NTOK/256, QKV_N/64), 256, 0, stream>>>(
      xw, wb_qkv, b_qkv, DIM, QKV_N, big, nullptr, nullptr);
  attn_wmma<<<NTOK/64, 512, 0, stream>>>(big, pb, att);
  gemm_wmma<EPI_PROJ_SCATTER><<<dim3(NTOK/256, DIM/64), 256, 0, stream>>>(
      att, wb_proj, b_proj, DIM, DIM, nullptr, x1, x);
  ln_kernel<false><<<NTOK/8, 256, 0, stream>>>(x1, g2, be2, att);
  gemm_wmma<EPI_BF16_GELU><<<dim3(NTOK/256, MLP_HID/64), 256, 0, stream>>>(
      att, wb_fc1, b_fc1, DIM, MLP_HID, big, nullptr, nullptr);
  gemm_wmma<EPI_F32_RES><<<dim3(NTOK/256, DIM/64), 256, 0, stream>>>(
      big, wb_fc2, b_fc2, MLP_HID, DIM, nullptr, out, x1);
}
